// MoEConformerLayer_27367531610288
// MI455X (gfx1250) — matrix operand order repaired and checked by reading, not compiled, not run on hardware
//
#include <hip/hip_runtime.h>
#include <hip/hip_bf16.h>

// MoE-Conformer layer for MI455X (gfx1250, wave32, WMMA).
// GEMMs: bf16 WMMA (v_wmma_f32_16x16x32_bf16), fp32 accumulation.
// Wave tile 32x64 (2 A-frags x 4 B-frags = 8 WMMA/K-step).
// Conv GEMM: block-shared B panel staged in LDS via global_load_async_to_lds_b128
// (ASYNCcnt) with double buffering.

typedef unsigned short u16;
typedef __bf16 v16bf __attribute__((ext_vector_type(16)));
typedef float  v8f   __attribute__((ext_vector_type(8)));

#define BB   2
#define SS   1024
#define DD   1024
#define HH   16
#define HDD  64
#define KK   31
#define GG   4
#define EE   2
#define HID  2048
#define NTOK (BB * SS)

__device__ __forceinline__ u16 f2bf(float f) {
  unsigned int u = __float_as_uint(f);
  u = (u + 0x7FFFu + ((u >> 16) & 1u)) >> 16;   // round-to-nearest-even
  return (u16)u;
}

union FragU { v16bf v; uint4 u[2]; };

__device__ __forceinline__ v16bf zero_frag() {
  FragU f;
  f.u[0] = make_uint4(0u, 0u, 0u, 0u);
  f.u[1] = make_uint4(0u, 0u, 0u, 0u);
  return f.v;
}

// A fragment (16x32 bf16, row-major A): lane l -> row l&15.
// lanes 0-15 hold K = k0..k0+15, lanes 16-31 hold K = k0+16..k0+31.
__device__ __forceinline__ v16bf load_a_frag(const u16* rowPtr, int k0, int lane) {
  FragU f;
  const u16* p = rowPtr + k0 + ((lane >> 4) << 4);
  f.u[0] = *(const uint4*)(p);
  f.u[1] = *(const uint4*)(p + 8);
  return f.v;
}

// B fragment (32x16 bf16) from B^T ([N][K] row-major): lane l -> col l&15,
// lanes 0-15 hold K = k0..k0+15, lanes 16-31 hold K = k0+16..k0+31.
__device__ __forceinline__ v16bf load_b_frag(const u16* colPtr, int k0, int lane) {
  FragU f;
  const u16* p = colPtr + k0 + ((lane >> 4) << 4);
  f.u[0] = *(const uint4*)(p);
  f.u[1] = *(const uint4*)(p + 8);
  return f.v;
}

#define WMMA_BF16(a, b, c) \
  __builtin_amdgcn_wmma_f32_16x16x32_bf16(false, (a), false, (b), (short)0, (c), false, false)

// 32x64 wave-tile step: 1 B fragment feeds both A fragments.
__device__ __forceinline__ void gemm_step(v8f acc[2][4], v16bf a0, v16bf a1,
                                          const u16* const* bCol, int k0, int lane) {
#pragma unroll
  for (int j = 0; j < 4; ++j) {
    v16bf bf = load_b_frag(bCol[j], k0, lane);
    acc[0][j] = WMMA_BF16(a0, bf, acc[0][j]);
    acc[1][j] = WMMA_BF16(a1, bf, acc[1][j]);
  }
}

__device__ __forceinline__ float gelu_exact(float v) {
  return 0.5f * v * (1.0f + erff(v * 0.70710678118654752f));
}

// Async copy 16B global -> LDS (CDNA5 ASYNCcnt path).
__device__ __forceinline__ void async_copy_b128(u16* lds_dst, const u16* gsrc) {
  unsigned lds_off = (unsigned)(uintptr_t)lds_dst;           // flat shared addr[31:0] = LDS offset
  unsigned long long ga = (unsigned long long)(uintptr_t)gsrc;
  asm volatile("global_load_async_to_lds_b128 %0, %1, off"
               :: "v"(lds_off), "v"(ga) : "memory");
}

__device__ __forceinline__ void wait_async() {
  asm volatile("s_wait_asynccnt 0" ::: "memory");
}

// ---------------------------------------------------------------------------
// Weight conversion / repack kernels
// ---------------------------------------------------------------------------

__global__ void k_f32_to_bf16(const float* __restrict__ src, u16* __restrict__ dst, int n) {
  int i = blockIdx.x * 256 + threadIdx.x;
  if (i < n) dst[i] = f2bf(src[i]);
}

// conv_w [O][I][K] -> Wt [O][K*D + I]  (B^T layout, K-dim = 31*1024)
__global__ void k_conv_repack(const float* __restrict__ cw, u16* __restrict__ Wt) {
  int idx = blockIdx.x * 256 + threadIdx.x;
  const int total = DD * DD * KK;
  if (idx >= total) return;
  int k  = idx % KK;
  int oi = idx / KK;
  int i  = oi % DD;
  int o  = oi / DD;
  Wt[(size_t)o * (KK * DD) + (size_t)k * DD + i] = f2bf(cw[idx]);
}

// batched transpose: src fp32 [batch][R][C] -> dst bf16 [batch][C][R]
__global__ void k_transpose_bf16(const float* __restrict__ src, u16* __restrict__ dst,
                                 int R, int C) {
  size_t base = (size_t)blockIdx.y * R * C;
  int idx = blockIdx.x * 256 + threadIdx.x;
  if (idx >= R * C) return;
  int r = idx / C, c = idx % C;
  dst[base + (size_t)c * R + r] = f2bf(src[base + (size_t)r * C + c]);
}

// ---------------------------------------------------------------------------
// LayerNorm: one block per row of 1024, fp32 in -> bf16 out
// ---------------------------------------------------------------------------
__global__ void __launch_bounds__(256)
k_layernorm(const float* __restrict__ x, const float* __restrict__ g,
            const float* __restrict__ b, u16* __restrict__ out) {
  __shared__ float sm[256];
  const int row = blockIdx.x;
  const float* xr = x + (size_t)row * DD;
  float v[4];
  float s = 0.f;
#pragma unroll
  for (int i = 0; i < 4; ++i) { v[i] = xr[threadIdx.x + i * 256]; s += v[i]; }
  sm[threadIdx.x] = s; __syncthreads();
  for (int t = 128; t > 0; t >>= 1) {
    if (threadIdx.x < t) sm[threadIdx.x] += sm[threadIdx.x + t];
    __syncthreads();
  }
  const float mu = sm[0] * (1.0f / DD);
  __syncthreads();
  float s2 = 0.f;
#pragma unroll
  for (int i = 0; i < 4; ++i) { float d = v[i] - mu; s2 += d * d; }
  sm[threadIdx.x] = s2; __syncthreads();
  for (int t = 128; t > 0; t >>= 1) {
    if (threadIdx.x < t) sm[threadIdx.x] += sm[threadIdx.x + t];
    __syncthreads();
  }
  const float rstd = rsqrtf(sm[0] * (1.0f / DD) + 1e-5f);
#pragma unroll
  for (int i = 0; i < 4; ++i) {
    int c = threadIdx.x + i * 256;
    out[(size_t)row * DD + c] = f2bf((v[i] - mu) * rstd * g[c] + b[c]);
  }
}

// ---------------------------------------------------------------------------
// Conv block as implicit GEMM: x1 = gelu(conv(LN(x)) + conv_b) + x
// Block tile: 256(M) x 64(N); 8 waves of 32x64. B panel (64x32 bf16 = 4KB)
// double-buffered in LDS via async copies, shared by all 8 waves.
// ---------------------------------------------------------------------------
__global__ void __launch_bounds__(256)
k_conv_gemm(const u16* __restrict__ hln, const u16* __restrict__ Wt,
            const float* __restrict__ conv_b, const float* __restrict__ xres,
            float* __restrict__ x1) {
  __shared__ u16 sB[2][64 * 32];
  const int lane = threadIdx.x & 31, wave = threadIdx.x >> 5;
  const int tmB = blockIdx.x >> 4;       // M block (256 rows), 0..7
  const int tn  = blockIdx.x & 15;       // N block (64 cols)
  const int ldb = KK * DD;

  // Cooperative B staging: 256 threads x one b128 each = 64 cols x 32 k.
  const int scol  = threadIdx.x >> 2;          // 0..63
  const int skoff = (threadIdx.x & 3) << 3;    // 0,8,16,24
  const u16* sgp = Wt + (size_t)(tn * 64 + scol) * ldb + skoff;
  u16* sdp[2] = { &sB[0][scol * 32 + skoff], &sB[1][scol * 32 + skoff] };

  // A rows for this wave (two 16-row sub-tiles)
  const int rA0 = tmB * 256 + wave * 32 + (lane & 15);
  const int rA1 = rA0 + 16;
  const int bI0 = rA0 >> 10, sI0 = rA0 & (SS - 1);
  const int bI1 = rA1 >> 10, sI1 = rA1 & (SS - 1);

  v8f acc[2][4] = {};

  // prologue: stage step 0
  async_copy_b128(sdp[0], sgp);
  wait_async();
  __syncthreads();

  const int TSTEPS = KK * (DD / 32);     // 992
  int ph = 0;
  for (int s = 0; s < TSTEPS; ++s) {
    if (s + 1 < TSTEPS) {                // prefetch next B panel into other buffer
      const int k1 = (s + 1) >> 5, kk1 = ((s + 1) & 31) << 5;
      async_copy_b128(sdp[ph ^ 1], sgp + (size_t)k1 * DD + kk1);
    }
    const int k = s >> 5, kk = (s & 31) << 5;
    const int sp0 = sI0 + k - KK / 2;
    const int sp1 = sI1 + k - KK / 2;
    const bool ok0 = (sp0 >= 0) && (sp0 < SS);
    const bool ok1 = (sp1 >= 0) && (sp1 < SS);
    v16bf a0 = ok0 ? load_a_frag(hln + (size_t)((bI0 << 10) + sp0) * DD, kk, lane) : zero_frag();
    v16bf a1 = ok1 ? load_a_frag(hln + (size_t)((bI1 << 10) + sp1) * DD, kk, lane) : zero_frag();
#pragma unroll
    for (int j = 0; j < 4; ++j) {
      const u16* bp = &sB[ph][(j * 16 + (lane & 15)) * 32 + ((lane >> 4) << 4)];
      FragU f;
      f.u[0] = *(const uint4*)(bp);
      f.u[1] = *(const uint4*)(bp + 8);
      acc[0][j] = WMMA_BF16(a0, f.v, acc[0][j]);
      acc[1][j] = WMMA_BF16(a1, f.v, acc[1][j]);
    }
    wait_async();
    __syncthreads();
    ph ^= 1;
  }

  const int mBase = tmB * 256 + wave * 32 + ((lane >> 4) << 3);
  const int nBase = tn * 64 + (lane & 15);
#pragma unroll
  for (int i = 0; i < 2; ++i)
#pragma unroll
    for (int j = 0; j < 4; ++j) {
      const int c = nBase + j * 16;
#pragma unroll
      for (int v = 0; v < 8; ++v) {
        const int m = mBase + i * 16 + v;
        float val = acc[i][j][v] + conv_b[c];
        val = gelu_exact(val);
        x1[(size_t)m * DD + c] = val + xres[(size_t)m * DD + c];
      }
    }
}

// ---------------------------------------------------------------------------
// QKV projection: [2048,1024] x in_w^T (+bias), repacked into Q/K/V^T bf16
// ---------------------------------------------------------------------------
__global__ void __launch_bounds__(256)
k_qkv_gemm(const u16* __restrict__ A, const u16* __restrict__ Bt,
           const float* __restrict__ bias,
           u16* __restrict__ qb, u16* __restrict__ kbuf, u16* __restrict__ vT) {
  const int lane = threadIdx.x & 31, wave = threadIdx.x >> 5;
  const int tid = blockIdx.x * 8 + wave;
  const int tiles_n = 3 * DD / 64;   // 48
  const int tm = tid / tiles_n, tn = tid % tiles_n;
  const u16* aRow0 = A + (size_t)(tm * 32 + (lane & 15)) * DD;
  const u16* aRow1 = aRow0 + (size_t)16 * DD;
  const u16* bCol[4];
#pragma unroll
  for (int j = 0; j < 4; ++j)
    bCol[j] = Bt + (size_t)(tn * 64 + j * 16 + (lane & 15)) * DD;
  v8f acc[2][4] = {};
  for (int k0 = 0; k0 < DD; k0 += 32) {
    v16bf a0 = load_a_frag(aRow0, k0, lane);
    v16bf a1 = load_a_frag(aRow1, k0, lane);
    gemm_step(acc, a0, a1, bCol, k0, lane);
  }
  const int mBase = tm * 32 + ((lane >> 4) << 3);
  const int nBase = tn * 64 + (lane & 15);
#pragma unroll
  for (int i = 0; i < 2; ++i)
#pragma unroll
    for (int j = 0; j < 4; ++j) {
      const int c = nBase + j * 16;
      const int part = c >> 10, hh = c & 1023, head = hh >> 6, hd = hh & 63;
#pragma unroll
      for (int v = 0; v < 8; ++v) {
        const int m = mBase + i * 16 + v;
        const int bI = m >> 10, sI = m & 1023;
        u16 bv = f2bf(acc[i][j][v] + bias[c]);
        if (part == 0)
          qb[((size_t)(bI * HH + head) * SS + sI) * HDD + hd] = bv;
        else if (part == 1)
          kbuf[((size_t)(bI * HH + head) * SS + sI) * HDD + hd] = bv;
        else
          vT[((size_t)(bI * HH + head) * HDD + hd) * SS + sI] = bv;
      }
    }
}

// ---------------------------------------------------------------------------
// scores = Q K^T / sqrt(64), per (b,h) slice (blockIdx.y = b*16+h)
// ---------------------------------------------------------------------------
__global__ void __launch_bounds__(256)
k_scores_gemm(const u16* __restrict__ qb, const u16* __restrict__ kbuf,
              float* __restrict__ scores) {
  const int z = blockIdx.y;
  const int lane = threadIdx.x & 31, wave = threadIdx.x >> 5;
  const int tid = blockIdx.x * 8 + wave;
  const int tiles_n = SS / 64;   // 16
  const int tm = tid / tiles_n, tn = tid % tiles_n;
  const u16* Aq = qb   + (size_t)z * SS * HDD;
  const u16* Bt = kbuf + (size_t)z * SS * HDD;
  const u16* aRow0 = Aq + (size_t)(tm * 32 + (lane & 15)) * HDD;
  const u16* aRow1 = aRow0 + (size_t)16 * HDD;
  const u16* bCol[4];
#pragma unroll
  for (int j = 0; j < 4; ++j)
    bCol[j] = Bt + (size_t)(tn * 64 + j * 16 + (lane & 15)) * HDD;
  v8f acc[2][4] = {};
#pragma unroll
  for (int k0 = 0; k0 < HDD; k0 += 32) {
    v16bf a0 = load_a_frag(aRow0, k0, lane);
    v16bf a1 = load_a_frag(aRow1, k0, lane);
    gemm_step(acc, a0, a1, bCol, k0, lane);
  }
  const int mBase = tm * 32 + ((lane >> 4) << 3);
  const int nBase = tn * 64 + (lane & 15);
#pragma unroll
  for (int i = 0; i < 2; ++i)
#pragma unroll
    for (int j = 0; j < 4; ++j) {
      const int c = nBase + j * 16;
#pragma unroll
      for (int v = 0; v < 8; ++v) {
        const int m = mBase + i * 16 + v;
        scores[(size_t)z * SS * SS + (size_t)m * SS + c] = acc[i][j][v] * 0.125f;
      }
    }
}

// ---------------------------------------------------------------------------
// Row softmax over 1024, fp32 in -> bf16 P out
// ---------------------------------------------------------------------------
__global__ void __launch_bounds__(256)
k_softmax(const float* __restrict__ S, u16* __restrict__ P) {
  __shared__ float sm[256];
  const size_t row = blockIdx.x;
  const float* sr = S + row * SS;
  float v[4];
  float mx = -3.4e38f;
#pragma unroll
  for (int i = 0; i < 4; ++i) { v[i] = sr[threadIdx.x + i * 256]; mx = fmaxf(mx, v[i]); }
  sm[threadIdx.x] = mx; __syncthreads();
  for (int t = 128; t > 0; t >>= 1) {
    if (threadIdx.x < t) sm[threadIdx.x] = fmaxf(sm[threadIdx.x], sm[threadIdx.x + t]);
    __syncthreads();
  }
  mx = sm[0]; __syncthreads();
  float s = 0.f;
#pragma unroll
  for (int i = 0; i < 4; ++i) { v[i] = __expf(v[i] - mx); s += v[i]; }
  sm[threadIdx.x] = s; __syncthreads();
  for (int t = 128; t > 0; t >>= 1) {
    if (threadIdx.x < t) sm[threadIdx.x] += sm[threadIdx.x + t];
    __syncthreads();
  }
  const float inv = 1.0f / sm[0];
#pragma unroll
  for (int i = 0; i < 4; ++i)
    P[row * SS + threadIdx.x + i * 256] = f2bf(v[i] * inv);
}

// ---------------------------------------------------------------------------
// ctx = P @ V, per (b,h); B^T = V^T [64][1024]; output packed [B,S,D] bf16
// ---------------------------------------------------------------------------
__global__ void __launch_bounds__(256)
k_ctx_gemm(const u16* __restrict__ P, const u16* __restrict__ vT,
           u16* __restrict__ ctxb) {
  const int z = blockIdx.y;                 // b*16 + head
  const int lane = threadIdx.x & 31, wave = threadIdx.x >> 5;
  const int tm = blockIdx.x * 8 + wave;     // 32 m-tiles of 32 rows, single 64-wide n tile
  const u16* Ap = P  + (size_t)z * SS * SS;
  const u16* Bt = vT + (size_t)z * HDD * SS;
  const u16* aRow0 = Ap + (size_t)(tm * 32 + (lane & 15)) * SS;
  const u16* aRow1 = aRow0 + (size_t)16 * SS;
  const u16* bCol[4];
#pragma unroll
  for (int j = 0; j < 4; ++j)
    bCol[j] = Bt + (size_t)(j * 16 + (lane & 15)) * SS;
  v8f acc[2][4] = {};
  for (int k0 = 0; k0 < SS; k0 += 32) {
    v16bf a0 = load_a_frag(aRow0, k0, lane);
    v16bf a1 = load_a_frag(aRow1, k0, lane);
    gemm_step(acc, a0, a1, bCol, k0, lane);
  }
  const int bI = z >> 4, head = z & 15;
  const int mBase = tm * 32 + ((lane >> 4) << 3);
  const int nBase = (lane & 15);
#pragma unroll
  for (int i = 0; i < 2; ++i)
#pragma unroll
    for (int j = 0; j < 4; ++j) {
      const int c = nBase + j * 16;             // 0..63 within head
#pragma unroll
      for (int v = 0; v < 8; ++v) {
        const int sI = mBase + i * 16 + v;
        ctxb[((size_t)bI * SS + sI) * DD + head * HDD + c] = f2bf(acc[i][j][v]);
      }
    }
}

// ---------------------------------------------------------------------------
// Out projection: x2 = ctx @ out_w^T + out_b + x1 (fp32 + bf16 copies)
// ---------------------------------------------------------------------------
__global__ void __launch_bounds__(256)
k_outproj_gemm(const u16* __restrict__ A, const u16* __restrict__ Bt,
               const float* __restrict__ bias, const float* __restrict__ res,
               float* __restrict__ x2, u16* __restrict__ x2b) {
  const int lane = threadIdx.x & 31, wave = threadIdx.x >> 5;
  const int tid = blockIdx.x * 8 + wave;
  const int tiles_n = DD / 64;
  const int tm = tid / tiles_n, tn = tid % tiles_n;
  const u16* aRow0 = A + (size_t)(tm * 32 + (lane & 15)) * DD;
  const u16* aRow1 = aRow0 + (size_t)16 * DD;
  const u16* bCol[4];
#pragma unroll
  for (int j = 0; j < 4; ++j)
    bCol[j] = Bt + (size_t)(tn * 64 + j * 16 + (lane & 15)) * DD;
  v8f acc[2][4] = {};
  for (int k0 = 0; k0 < DD; k0 += 32) {
    v16bf a0 = load_a_frag(aRow0, k0, lane);
    v16bf a1 = load_a_frag(aRow1, k0, lane);
    gemm_step(acc, a0, a1, bCol, k0, lane);
  }
  const int mBase = tm * 32 + ((lane >> 4) << 3);
  const int nBase = tn * 64 + (lane & 15);
#pragma unroll
  for (int i = 0; i < 2; ++i)
#pragma unroll
    for (int j = 0; j < 4; ++j) {
      const int c = nBase + j * 16;
#pragma unroll
      for (int v = 0; v < 8; ++v) {
        const int m = mBase + i * 16 + v;
        float val = acc[i][j][v] + bias[c] + res[(size_t)m * DD + c];
        x2[(size_t)m * DD + c]  = val;
        x2b[(size_t)m * DD + c] = f2bf(val);
      }
    }
}

// ---------------------------------------------------------------------------
// MoE: gather tokens by group, then run only the selected group's experts
// ---------------------------------------------------------------------------
__global__ void k_zero_cnt(int* cnt) {
  if (threadIdx.x < GG) cnt[threadIdx.x] = 0;
}

__global__ void k_gather(const int* __restrict__ gid, int* __restrict__ cnt,
                         int* __restrict__ list) {
  int n = blockIdx.x * 256 + threadIdx.x;
  if (n >= NTOK) return;
  int g = gid[n];
  int pos = atomicAdd(&cnt[g], 1);
  list[g * NTOK + pos] = n;
}

// h1[e][token][HID] = gelu(x2b[token] @ e_w1[g][e] + e_b1[g][e]) for token's group
__global__ void __launch_bounds__(256)
k_moe_mlp1(const u16* __restrict__ x2b, const u16* __restrict__ e1t,
           const float* __restrict__ e_b1, const int* __restrict__ cnt,
           const int* __restrict__ list, u16* __restrict__ h1) {
  const int ge = blockIdx.y, g = ge >> 1;
  const int Ng = cnt[g];
  const int lane = threadIdx.x & 31, wave = threadIdx.x >> 5;
  const int tid = blockIdx.x * 8 + wave;
  const int tiles_n = HID / 64;   // 32
  const int tm = tid / tiles_n, tn = tid % tiles_n;
  if (tm * 32 >= Ng) return;
  const int e = ge & 1;
  const int rA0 = tm * 32 + (lane & 15);
  const int rA1 = rA0 + 16;
  const int tA0 = list[g * NTOK + (rA0 < Ng ? rA0 : Ng - 1)];
  const int tA1 = list[g * NTOK + (rA1 < Ng ? rA1 : Ng - 1)];
  const u16* aRow0 = x2b + (size_t)tA0 * DD;
  const u16* aRow1 = x2b + (size_t)tA1 * DD;
  const u16* Bt = e1t + (size_t)ge * HID * DD;
  const u16* bCol[4];
#pragma unroll
  for (int j = 0; j < 4; ++j)
    bCol[j] = Bt + (size_t)(tn * 64 + j * 16 + (lane & 15)) * DD;
  v8f acc[2][4] = {};
  for (int k0 = 0; k0 < DD; k0 += 32) {
    v16bf a0 = load_a_frag(aRow0, k0, lane);
    v16bf a1 = load_a_frag(aRow1, k0, lane);
    gemm_step(acc, a0, a1, bCol, k0, lane);
  }
  const int mBase = tm * 32 + ((lane >> 4) << 3);
  const int nBase = tn * 64 + (lane & 15);
#pragma unroll
  for (int i = 0; i < 2; ++i)
#pragma unroll
    for (int j = 0; j < 4; ++j) {
      const int c = nBase + j * 16;
#pragma unroll
      for (int v = 0; v < 8; ++v) {
        const int m = mBase + i * 16 + v;
        if (m >= Ng) continue;
        const int t = list[g * NTOK + m];
        float val = gelu_exact(acc[i][j][v] + e_b1[(size_t)ge * HID + c]);
        h1[((size_t)e * NTOK + t) * HID + c] = f2bf(val);
      }
    }
}

// out[token] = mean_e(h1[e][token] @ e_w2[g][e] + e_b2[g][e]) + x2[token]
__global__ void __launch_bounds__(256)
k_moe_mlp2(const u16* __restrict__ h1, const u16* __restrict__ e2t,
           const float* __restrict__ e_b2, const int* __restrict__ cnt,
           const int* __restrict__ list, const float* __restrict__ res,
           float* __restrict__ out) {
  const int g = blockIdx.y;
  const int Ng = cnt[g];
  const int lane = threadIdx.x & 31, wave = threadIdx.x >> 5;
  const int tid = blockIdx.x * 8 + wave;
  const int tiles_n = DD / 64;    // 16
  const int tm = tid / tiles_n, tn = tid % tiles_n;
  if (tm * 32 >= Ng) return;
  const int rA0 = tm * 32 + (lane & 15);
  const int rA1 = rA0 + 16;
  const int tA0 = list[g * NTOK + (rA0 < Ng ? rA0 : Ng - 1)];
  const int tA1 = list[g * NTOK + (rA1 < Ng ? rA1 : Ng - 1)];
  v8f acc[2][4] = {};
  for (int e = 0; e < EE; ++e) {
    const u16* aRow0 = h1 + ((size_t)e * NTOK + tA0) * HID;
    const u16* aRow1 = h1 + ((size_t)e * NTOK + tA1) * HID;
    const u16* BtE = e2t + (size_t)(g * EE + e) * DD * HID;
    const u16* bCol[4];
#pragma unroll
    for (int j = 0; j < 4; ++j)
      bCol[j] = BtE + (size_t)(tn * 64 + j * 16 + (lane & 15)) * HID;
    for (int k0 = 0; k0 < HID; k0 += 32) {
      v16bf a0 = load_a_frag(aRow0, k0, lane);
      v16bf a1 = load_a_frag(aRow1, k0, lane);
      gemm_step(acc, a0, a1, bCol, k0, lane);
    }
  }
  const int mBase = tm * 32 + ((lane >> 4) << 3);
  const int nBase = tn * 64 + (lane & 15);
#pragma unroll
  for (int i = 0; i < 2; ++i)
#pragma unroll
    for (int j = 0; j < 4; ++j) {
      const int c = nBase + j * 16;
      const float bias =
          0.5f * (e_b2[(size_t)(g * EE + 0) * DD + c] + e_b2[(size_t)(g * EE + 1) * DD + c]);
#pragma unroll
      for (int v = 0; v < 8; ++v) {
        const int m = mBase + i * 16 + v;
        if (m >= Ng) continue;
        const int t = list[g * NTOK + m];
        out[(size_t)t * DD + c] = 0.5f * acc[i][j][v] + bias + res[(size_t)t * DD + c];
      }
    }
}

// ---------------------------------------------------------------------------
// Host launcher
// ---------------------------------------------------------------------------
extern "C" void kernel_launch(void* const* d_in, const int* in_sizes, int n_in,
                              void* d_out, int out_size, void* d_ws, size_t ws_size,
                              hipStream_t stream) {
  (void)in_sizes; (void)n_in; (void)out_size; (void)ws_size;
  const float* x      = (const float*)d_in[0];
  const int*   gid    = (const int*)  d_in[1];
  const float* cn_g   = (const float*)d_in[2];
  const float* cn_b   = (const float*)d_in[3];
  const float* conv_w = (const float*)d_in[4];
  const float* conv_b = (const float*)d_in[5];
  const float* an_g   = (const float*)d_in[6];
  const float* an_b   = (const float*)d_in[7];
  const float* in_w   = (const float*)d_in[8];
  const float* in_b   = (const float*)d_in[9];
  const float* out_w  = (const float*)d_in[10];
  const float* out_b  = (const float*)d_in[11];
  const float* e_w1   = (const float*)d_in[12];
  const float* e_b1   = (const float*)d_in[13];
  const float* e_w2   = (const float*)d_in[14];
  const float* e_b2   = (const float*)d_in[15];
  float* out = (float*)d_out;

  char* p = (char*)d_ws;
  auto alloc = [&](size_t bytes) -> char* {
    char* r = p;
    p += (bytes + 255) & ~(size_t)255;
    return r;
  };
  u16*   hln    = (u16*)  alloc((size_t)NTOK * DD * 2);
  u16*   WtConv = (u16*)  alloc((size_t)DD * KK * DD * 2);
  float* x1     = (float*)alloc((size_t)NTOK * DD * 4);
  u16*   h2     = (u16*)  alloc((size_t)NTOK * DD * 2);
  u16*   inwB   = (u16*)  alloc((size_t)3 * DD * DD * 2);
  u16*   outwB  = (u16*)  alloc((size_t)DD * DD * 2);
  u16*   qb     = (u16*)  alloc((size_t)BB * HH * SS * HDD * 2);
  u16*   kbuf   = (u16*)  alloc((size_t)BB * HH * SS * HDD * 2);
  u16*   vT     = (u16*)  alloc((size_t)BB * HH * HDD * SS * 2);
  float* scores = (float*)alloc((size_t)BB * HH * SS * SS * 4);
  u16*   Pm     = (u16*)  alloc((size_t)BB * HH * SS * SS * 2);
  u16*   ctxb   = (u16*)  alloc((size_t)NTOK * DD * 2);
  float* x2     = (float*)alloc((size_t)NTOK * DD * 4);
  u16*   x2b    = (u16*)  alloc((size_t)NTOK * DD * 2);
  u16*   e1t    = (u16*)  alloc((size_t)GG * EE * HID * DD * 2);
  u16*   e2t    = (u16*)  alloc((size_t)GG * EE * DD * HID * 2);
  u16*   h1     = (u16*)  alloc((size_t)EE * NTOK * HID * 2);
  int*   cnt    = (int*)  alloc(GG * 4);
  int*   list   = (int*)  alloc((size_t)GG * NTOK * 4);

  // --- weight conversions (bf16 / repack) ---
  {
    int n = DD * DD * KK;
    k_conv_repack<<<(n + 255) / 256, 256, 0, stream>>>(conv_w, WtConv);
  }
  {
    int n = 3 * DD * DD;
    k_f32_to_bf16<<<(n + 255) / 256, 256, 0, stream>>>(in_w, inwB, n);
  }
  {
    int n = DD * DD;
    k_f32_to_bf16<<<(n + 255) / 256, 256, 0, stream>>>(out_w, outwB, n);
  }
  {
    dim3 grid((DD * HID + 255) / 256, GG * EE);
    k_transpose_bf16<<<grid, 256, 0, stream>>>(e_w1, e1t, DD, HID);   // -> [ge][HID][D]
    k_transpose_bf16<<<grid, 256, 0, stream>>>(e_w2, e2t, HID, DD);   // -> [ge][D][HID]
  }

  // --- conv block ---
  k_layernorm<<<NTOK, 256, 0, stream>>>(x, cn_g, cn_b, hln);
  k_conv_gemm<<<(NTOK / 256) * (DD / 64), 256, 0, stream>>>(hln, WtConv, conv_b, x, x1);

  // --- attention block ---
  k_layernorm<<<NTOK, 256, 0, stream>>>(x1, an_g, an_b, h2);
  k_qkv_gemm<<<(NTOK / 32) * (3 * DD / 64) / 8, 256, 0, stream>>>(h2, inwB, in_b, qb, kbuf, vT);
  {
    dim3 grid((SS / 32) * (SS / 64) / 8, BB * HH);
    k_scores_gemm<<<grid, 256, 0, stream>>>(qb, kbuf, scores);
  }
  k_softmax<<<BB * HH * SS, 256, 0, stream>>>(scores, Pm);
  {
    dim3 grid((SS / 32) / 8, BB * HH);
    k_ctx_gemm<<<grid, 256, 0, stream>>>(Pm, vT, ctxb);
  }
  k_outproj_gemm<<<(NTOK / 32) * (DD / 64) / 8, 256, 0, stream>>>(ctxb, outwB, out_b, x1, x2, x2b);

  // --- group-routed MoE (only the selected group's experts are computed) ---
  k_zero_cnt<<<1, 64, 0, stream>>>(cnt);
  k_gather<<<(NTOK + 255) / 256, 256, 0, stream>>>(gid, cnt, list);
  {
    dim3 grid((NTOK / 32) * (HID / 64) / 8, GG * EE);   // worst case: all tokens one group
    k_moe_mlp1<<<grid, 256, 0, stream>>>(x2b, e1t, e_b1, cnt, list, h1);
  }
  {
    dim3 grid((NTOK / 32) * (DD / 64) / 8, GG);
    k_moe_mlp2<<<grid, 256, 0, stream>>>(h1, e2t, e_b2, cnt, list, x2, out);
  }
}
